// HF_T5_Partial_38594576122356
// MI455X (gfx1250) — matrix operand order repaired and checked by reading, not compiled, not run on hardware
//
#include <hip/hip_runtime.h>

// ---------------------------------------------------------------------------
// Types for CDNA5 WMMA (wave32): v_wmma_f32_16x16x32_bf16
// ---------------------------------------------------------------------------
typedef __bf16 bf16;
typedef __attribute__((ext_vector_type(8)))  __bf16 bf16x8;
typedef __attribute__((ext_vector_type(16))) __bf16 v16bf;
typedef __attribute__((ext_vector_type(8)))  float  v8f;

union V16U { v16bf v; bf16x8 h[2]; };

static __device__ inline v16bf pack16(bf16x8 lo, bf16x8 hi) {
  V16U u; u.h[0] = lo; u.h[1] = hi; return u.v;
}

static __device__ inline v8f wmma_bf16(v16bf a, v16bf b, v8f c) {
  return __builtin_amdgcn_wmma_f32_16x16x32_bf16(false, a, false, b,
                                                 (short)0, c, false, false);
}

// Async global->LDS copy, 16B per lane, tracked by ASYNCcnt.
// lds_off: wave-relative LDS byte address (low 32 bits of generic pointer).
static __device__ inline void async_load_b128(unsigned lds_off, const void* gaddr) {
  asm volatile("global_load_async_to_lds_b128 %0, %1, off"
               :: "v"(lds_off), "v"(gaddr) : "memory");
}
static __device__ inline unsigned lds_addr(const void* p) {
  return (unsigned)(unsigned long long)p;   // LDS aperture: addr[31:0] = LDS offset
}

// Problem constants
#define BB   8
#define SS   1024
#define DD   512
#define HH   8
#define HDD  64
#define DFFF 2048
#define MM   (BB * SS)   // 8192 tokens

// ---------------------------------------------------------------------------
// fp32 -> bf16 conversion (weights)
// ---------------------------------------------------------------------------
__global__ void __launch_bounds__(256) cvt_bf16_kernel(const float* __restrict__ in,
                                                       bf16* __restrict__ out, int n) {
  int i = blockIdx.x * 256 + threadIdx.x;
  if (i < n) out[i] = (bf16)in[i];
}

// ---------------------------------------------------------------------------
// RMSNorm: x [rows,512] fp32, w [512] fp32 -> out bf16 [rows,512]
// ---------------------------------------------------------------------------
__global__ void __launch_bounds__(256) rmsnorm_kernel(const float* __restrict__ x,
                                                      const float* __restrict__ w,
                                                      bf16* __restrict__ out) {
  __shared__ float red[256];
  const int row = blockIdx.x, tid = threadIdx.x;
  const float* xr = x + (size_t)row * DD;
  float v0 = xr[tid], v1 = xr[tid + 256];
  red[tid] = v0 * v0 + v1 * v1;
  __syncthreads();
  for (int s = 128; s > 0; s >>= 1) {
    if (tid < s) red[tid] += red[tid + s];
    __syncthreads();
  }
  float scale = rsqrtf(red[0] * (1.0f / DD) + 1e-6f);
  out[(size_t)row * DD + tid]       = (bf16)(w[tid] * v0 * scale);
  out[(size_t)row * DD + tid + 256] = (bf16)(w[tid + 256] * v1 * scale);
}

// ---------------------------------------------------------------------------
// bf16 WMMA GEMM: C[M,N] = A[M,K] * B[N,K]^T   (nn.Linear convention)
// Block tile 128x64, 8 waves, each wave -> 32x32 (four f32 16x16 accumulators).
// K staged 32 deep, double-buffered in LDS via GLOBAL_LOAD_ASYNC_TO_LDS_B128
// (ASYNCcnt pipelining: next stage issued before computing current stage).
// ---------------------------------------------------------------------------
enum EpMode { EP_QK = 0, EP_VT = 1, EP_RELU = 2, EP_RESID = 3 };

template <int MODE>
__global__ void __launch_bounds__(256)
gemm_bf16_wmma(const bf16* __restrict__ A, const bf16* __restrict__ Bm,
               void* __restrict__ Cout, const float* __restrict__ resid,
               int Ndim, int Kdim) {
  __shared__ bf16 As[2][128][32];   // 16 KB
  __shared__ bf16 Bs[2][64][32];    //  8 KB

  const int tid  = threadIdx.x;
  const int lane = tid & 31, wave = tid >> 5;
  const int lrow = lane & 15, lhi = lane >> 4;
  const int m0 = blockIdx.y * 128, n0 = blockIdx.x * 64;
  const int rt = wave >> 1;   // 0..3 : 32-row slice
  const int ct = wave & 1;    // 0..1 : 32-col slice

  v8f acc00 = {0.f,0.f,0.f,0.f,0.f,0.f,0.f,0.f};
  v8f acc01 = acc00, acc10 = acc00, acc11 = acc00;

  // Stage-loader mapping: thread t owns A rows (t>>2) and (t>>2)+64, B row (t>>2),
  // columns (t&3)*8 .. +7  => 16B per async op, 3 ops/thread/stage.
  const int ldr  = tid >> 2;          // 0..63
  const int ldc  = (tid & 3) * 8;     // 0,8,16,24
  const bf16* gA0 = A  + (size_t)(m0 + ldr)      * Kdim + ldc;
  const bf16* gA1 = A  + (size_t)(m0 + ldr + 64) * Kdim + ldc;
  const bf16* gB  = Bm + (size_t)(n0 + ldr)      * Kdim + ldc;
  unsigned lA0[2], lA1[2], lB[2];
  for (int b = 0; b < 2; ++b) {
    lA0[b] = lds_addr(&As[b][ldr][ldc]);
    lA1[b] = lds_addr(&As[b][ldr + 64][ldc]);
    lB[b]  = lds_addr(&Bs[b][ldr][ldc]);
  }

  // Prologue: issue stage 0 into buffer 0.
  async_load_b128(lA0[0], gA0);
  async_load_b128(lA1[0], gA1);
  async_load_b128(lB[0],  gB);

  const int nstages = Kdim >> 5;
  for (int i = 0; i < nstages; ++i) {
    const int buf = i & 1;
    if (i + 1 < nstages) {
      const size_t koff = (size_t)(i + 1) * 32;
      async_load_b128(lA0[buf ^ 1], gA0 + koff);
      async_load_b128(lA1[buf ^ 1], gA1 + koff);
      async_load_b128(lB[buf ^ 1],  gB  + koff);
      // leave the 3 just-issued ops outstanding; stage i's ops are done
      asm volatile("s_wait_asynccnt 0x3" ::: "memory");
    } else {
      asm volatile("s_wait_asynccnt 0x0" ::: "memory");
    }
    __syncthreads();   // stage i visible to all waves

    // A fragments (16x32): lane = row lrow, K chunks {lhi*8..} / {16+lhi*8..}
    v16bf a0 = pack16(*reinterpret_cast<const bf16x8*>(&As[buf][rt*32 + lrow][lhi*8]),
                      *reinterpret_cast<const bf16x8*>(&As[buf][rt*32 + lrow][16 + lhi*8]));
    v16bf a1 = pack16(*reinterpret_cast<const bf16x8*>(&As[buf][rt*32 + 16 + lrow][lhi*8]),
                      *reinterpret_cast<const bf16x8*>(&As[buf][rt*32 + 16 + lrow][16 + lhi*8]));
    // B fragments (32x16): lane = column lrow, K chunks {lhi*8..} / {16+lhi*8..}
    v16bf b0 = pack16(*reinterpret_cast<const bf16x8*>(&Bs[buf][ct*32 + lrow][lhi*8]),
                      *reinterpret_cast<const bf16x8*>(&Bs[buf][ct*32 + lrow][16 + lhi*8]));
    v16bf b1 = pack16(*reinterpret_cast<const bf16x8*>(&Bs[buf][ct*32 + 16 + lrow][lhi*8]),
                      *reinterpret_cast<const bf16x8*>(&Bs[buf][ct*32 + 16 + lrow][16 + lhi*8]));

    acc00 = wmma_bf16(a0, b0, acc00);
    acc01 = wmma_bf16(a0, b1, acc01);
    acc10 = wmma_bf16(a1, b0, acc10);
    acc11 = wmma_bf16(a1, b1, acc11);

    __syncthreads();   // all waves done reading buf before it is overwritten
  }

  // Epilogue. C layout: element r of acc <-> M = r + 8*lhi, N = lrow (+tile)
  for (int rr = 0; rr < 2; ++rr) {
    for (int sub = 0; sub < 2; ++sub) {
      v8f acc = rr ? (sub ? acc11 : acc10) : (sub ? acc01 : acc00);
      for (int r = 0; r < 8; ++r) {
        int m = m0 + rt * 32 + rr * 16 + r + lhi * 8;
        int n = n0 + ct * 32 + sub * 16 + lrow;
        float val = acc[r];
        if (MODE == EP_QK) {            // store as [B,H,S,HD] bf16
          int bi = m >> 10, s = m & 1023, hh = n >> 6, d = n & 63;
          ((bf16*)Cout)[((((size_t)bi * HH + hh) << 10) + s) * HDD + d] = (bf16)val;
        } else if (MODE == EP_VT) {     // store as [B,H,HD,S] bf16 (pre-transposed V)
          int bi = m >> 10, s = m & 1023, hh = n >> 6, d = n & 63;
          ((bf16*)Cout)[(((size_t)bi * HH + hh) * HDD + d) * SS + s] = (bf16)val;
        } else if (MODE == EP_RELU) {   // relu, bf16 row-major [M,N]
          ((bf16*)Cout)[(size_t)m * Ndim + n] = (bf16)fmaxf(val, 0.0f);
        } else {                        // + residual, fp32 row-major [M,N]
          ((float*)Cout)[(size_t)m * Ndim + n] = val + resid[(size_t)m * Ndim + n];
        }
      }
    }
  }
}

// ---------------------------------------------------------------------------
// Attention: one block per (b, h, 16-query tile).
// q,k : [B,H,S,HD] bf16.  vT : [B,H,HD,S] bf16.  bias : [B,H,S,S] fp32.
// ctx : [B,S,H,HD] bf16 (== [M, D] row-major for the output projection).
// ---------------------------------------------------------------------------
__global__ void __launch_bounds__(256)
attn_kernel(const bf16* __restrict__ q, const bf16* __restrict__ k,
            const bf16* __restrict__ vT, const float* __restrict__ bias,
            bf16* __restrict__ ctx) {
  __shared__ bf16  sc[16][SS];   // scores, then (in place) unnormalized probs
  __shared__ float red[16][16];
  __shared__ float rowmax[16];
  __shared__ float rowsum[16];

  const int tid  = threadIdx.x;
  const int lane = tid & 31, wave = tid >> 5;
  const int lrow = lane & 15, lhi = lane >> 4;

  const int wg = blockIdx.x;
  const int qt = wg & 63, hh = (wg >> 6) & 7, bi = wg >> 9;
  const int q0 = qt * 16;

  const size_t head = ((size_t)bi * HH + hh) * (size_t)SS * HDD;
  const bf16*  qh = q  + head;
  const bf16*  kh = k  + head;
  const bf16*  vh = vT + head;
  const float* biash = bias + (((size_t)bi * HH + hh) << 20);

  // Q fragments for rows q0..q0+15 (two k-steps over HD=64)
  const bf16* qrow = qh + (size_t)(q0 + lrow) * HDD;
  v16bf qa0 = pack16(*reinterpret_cast<const bf16x8*>(qrow + lhi * 8),
                     *reinterpret_cast<const bf16x8*>(qrow + 16 + lhi * 8));
  v16bf qa1 = pack16(*reinterpret_cast<const bf16x8*>(qrow + 32 + lhi * 8),
                     *reinterpret_cast<const bf16x8*>(qrow + 48 + lhi * 8));

  // ---- Phase 1: scores = Q K^T + bias ----
  for (int t = 0; t < 8; ++t) {
    const int kb = wave * 128 + t * 16;
    const bf16* krow = kh + (size_t)(kb + lrow) * HDD;
    v16bf kb0 = pack16(*reinterpret_cast<const bf16x8*>(krow + lhi * 8),
                       *reinterpret_cast<const bf16x8*>(krow + 16 + lhi * 8));
    v16bf kb1 = pack16(*reinterpret_cast<const bf16x8*>(krow + 32 + lhi * 8),
                       *reinterpret_cast<const bf16x8*>(krow + 48 + lhi * 8));
    v8f acc = {0.f,0.f,0.f,0.f,0.f,0.f,0.f,0.f};
    acc = wmma_bf16(qa0, kb0, acc);
    acc = wmma_bf16(qa1, kb1, acc);
    for (int r = 0; r < 8; ++r) {
      int m = r + lhi * 8;
      int n = kb + lrow;
      float sv = acc[r] + biash[(size_t)(q0 + m) * SS + n];
      sc[m][n] = (bf16)sv;
    }
  }
  __syncthreads();

  // ---- Phase 2: softmax over each of the 16 rows (1024 keys) ----
  {
    const int r = tid >> 4, cj = tid & 15;
    float mx = -3.0e38f;
    for (int j = 0; j < 64; ++j) {
      float s = (float)sc[r][cj + 16 * j];
      mx = fmaxf(mx, s);
    }
    red[r][cj] = mx;
    __syncthreads();
    if (cj == 0) {
      float m2 = red[r][0];
      for (int i = 1; i < 16; ++i) m2 = fmaxf(m2, red[r][i]);
      rowmax[r] = m2;
    }
    __syncthreads();
    const float rm = rowmax[r];
    float ps = 0.f;
    for (int j = 0; j < 64; ++j) {
      int c = cj + 16 * j;
      float e = __expf((float)sc[r][c] - rm);
      sc[r][c] = (bf16)e;                  // in place: element owned by this thread
      ps += e;
    }
    red[r][cj] = ps;
    __syncthreads();
    if (cj == 0) {
      float s2 = 0.f;
      for (int i = 0; i < 16; ++i) s2 += red[r][i];
      rowsum[r] = s2;
    }
    __syncthreads();
  }

  // ---- Phase 3: ctx = P V (waves 0..3, one 16-wide d-slice each) ----
  if (wave < 4) {
    const int d0 = wave * 16;
    v8f acc = {0.f,0.f,0.f,0.f,0.f,0.f,0.f,0.f};
    for (int kt = 0; kt < SS / 32; ++kt) {
      v16bf pa = pack16(*reinterpret_cast<const bf16x8*>(&sc[lrow][kt * 32 + lhi * 8]),
                        *reinterpret_cast<const bf16x8*>(&sc[lrow][kt * 32 + 16 + lhi * 8]));
      const bf16* vrow = vh + (size_t)(d0 + lrow) * SS + kt * 32;
      v16bf vb = pack16(*reinterpret_cast<const bf16x8*>(vrow + lhi * 8),
                        *reinterpret_cast<const bf16x8*>(vrow + 16 + lhi * 8));
      acc = wmma_bf16(pa, vb, acc);
    }
    for (int r = 0; r < 8; ++r) {
      int m = r + lhi * 8;
      int d = d0 + lrow;
      float val = acc[r] / rowsum[m];
      ctx[((((size_t)bi << 10) + q0 + m) * HH + hh) * HDD + d] = (bf16)val;
    }
  }
}

// ---------------------------------------------------------------------------
// Host launcher
// ---------------------------------------------------------------------------
extern "C" void kernel_launch(void* const* d_in, const int* in_sizes, int n_in,
                              void* d_out, int out_size, void* d_ws, size_t ws_size,
                              hipStream_t stream) {
  (void)in_sizes; (void)n_in; (void)out_size; (void)ws_size;
  const float* Wk   = (const float*)d_in[0];  // [D,D]
  const float* Wo   = (const float*)d_in[1];  // [D,D]
  const float* Wq   = (const float*)d_in[2];  // [D,D]
  const float* Wv   = (const float*)d_in[3];  // [D,D]
  const float* g1   = (const float*)d_in[4];  // [D]
  const float* Wi   = (const float*)d_in[5];  // [DFF,D]
  const float* Wo2  = (const float*)d_in[6];  // [D,DFF]
  const float* g2   = (const float*)d_in[7];  // [D]
  const float* X    = (const float*)d_in[8];  // [B,S,D]
  const float* Bias = (const float*)d_in[9];  // [B,H,S,S]
  float* Out = (float*)d_out;

  char* ws = (char*)d_ws;
  size_t off = 0;
  auto alloc = [&](size_t bytes) -> void* {
    void* p = ws + off;
    off += (bytes + 255) & ~(size_t)255;
    return p;
  };
  bf16* h    = (bf16*)alloc((size_t)MM * DD * 2);
  bf16* wqb  = (bf16*)alloc((size_t)DD * DD * 2);
  bf16* wkb  = (bf16*)alloc((size_t)DD * DD * 2);
  bf16* wvb  = (bf16*)alloc((size_t)DD * DD * 2);
  bf16* wob  = (bf16*)alloc((size_t)DD * DD * 2);
  bf16* wib  = (bf16*)alloc((size_t)DFFF * DD * 2);
  bf16* wo2b = (bf16*)alloc((size_t)DD * DFFF * 2);
  bf16* qb   = (bf16*)alloc((size_t)MM * DD * 2);
  bf16* kb   = (bf16*)alloc((size_t)MM * DD * 2);
  bf16* vtb  = (bf16*)alloc((size_t)MM * DD * 2);
  bf16* ctxb = (bf16*)alloc((size_t)MM * DD * 2);
  float* yb  = (float*)alloc((size_t)MM * DD * 4);
  bf16* h2   = (bf16*)alloc((size_t)MM * DD * 2);
  bf16* ffb  = (bf16*)alloc((size_t)MM * DFFF * 2);

  // Weight conversion fp32 -> bf16
  const int nDD  = DD * DD;        // 262144
  const int nDFF = DFFF * DD;      // 1048576
  cvt_bf16_kernel<<<(nDD + 255) / 256, 256, 0, stream>>>(Wq,  wqb,  nDD);
  cvt_bf16_kernel<<<(nDD + 255) / 256, 256, 0, stream>>>(Wk,  wkb,  nDD);
  cvt_bf16_kernel<<<(nDD + 255) / 256, 256, 0, stream>>>(Wv,  wvb,  nDD);
  cvt_bf16_kernel<<<(nDD + 255) / 256, 256, 0, stream>>>(Wo,  wob,  nDD);
  cvt_bf16_kernel<<<(nDFF + 255) / 256, 256, 0, stream>>>(Wi,  wib,  nDFF);
  cvt_bf16_kernel<<<(nDFF + 255) / 256, 256, 0, stream>>>(Wo2, wo2b, nDFF);

  // RMSNorm 1
  rmsnorm_kernel<<<MM, 256, 0, stream>>>(X, g1, h);

  // QKV projections: block tile 128x64 -> grid (N/64, M/128)
  dim3 gProj(DD / 64, MM / 128);
  gemm_bf16_wmma<EP_QK><<<gProj, 256, 0, stream>>>(h, wqb, qb,  nullptr, DD, DD);
  gemm_bf16_wmma<EP_QK><<<gProj, 256, 0, stream>>>(h, wkb, kb,  nullptr, DD, DD);
  gemm_bf16_wmma<EP_VT><<<gProj, 256, 0, stream>>>(h, wvb, vtb, nullptr, DD, DD);

  // Attention (B*H*(S/16) = 4096 blocks)
  attn_kernel<<<BB * HH * (SS / 16), 256, 0, stream>>>(qb, kb, vtb, Bias, ctxb);

  // Output projection + residual -> y (fp32)
  gemm_bf16_wmma<EP_RESID><<<gProj, 256, 0, stream>>>(ctxb, wob, yb, X, DD, DD);

  // RMSNorm 2
  rmsnorm_kernel<<<MM, 256, 0, stream>>>(yb, g2, h2);

  // FFN up (relu) : N = 2048 -> grid.x = 32
  dim3 gUp(DFFF / 64, MM / 128);
  gemm_bf16_wmma<EP_RELU><<<gUp, 256, 0, stream>>>(h2, wib, ffb, nullptr, DFFF, DD);

  // FFN down + residual -> final output (fp32)
  gemm_bf16_wmma<EP_RESID><<<gProj, 256, 0, stream>>>(ffb, wo2b, Out, yb, DD, DFFF);
}
